// FastGRNNCUDACell_29222957482088
// MI455X (gfx1250) — matrix operand order repaired and checked by reading, not compiled, not run on hardware
//
#include <hip/hip_runtime.h>
#include <hip/hip_bf16.h>

#define BM 128
#define BN 128
#define BK 32

typedef __attribute__((ext_vector_type(16))) __bf16 v16bf;
typedef __attribute__((ext_vector_type(8)))  float  v8f;

// ---------------------------------------------------------------------------
// LDS fragment loaders following CDNA5 ISA 7.12.2 VGPR layouts (wave32).
__device__ __forceinline__ v16bf frag_a(const __bf16* S, int m0, int lane) {
  const int m = m0 + (lane & 15);
  const int g = (lane >> 4) << 3;            // 0 or 8
  union { v16bf v; uint4 q[2]; } u;
  u.q[0] = *(const uint4*)(S + m * BK + g);
  u.q[1] = *(const uint4*)(S + m * BK + g + 16);
  return u.v;
}

__device__ __forceinline__ v16bf frag_b(const __bf16* S, int n0, int lane) {
  const int n  = n0 + (lane & 15);
  const int kb = (lane >> 4) << 3;           // 0 or 8
  union { v16bf v; uint4 q[2]; } u;
  u.q[0] = *(const uint4*)(S + n * BK + kb);
  u.q[1] = *(const uint4*)(S + n * BK + kb + 16);
  return u.v;
}

// ---------------------------------------------------------------------------
// Split staging: load phase (8 independent float4 -> all global loads in
// flight at once) and commit phase (cvt_pk_bf16 + ds_store) so the compiler
// can overlap HBM latency with the WMMA pipe instead of wait-per-chunk.
struct StageRegs { float4 v[4]; };

__device__ __forceinline__ void stage_load(const float* __restrict__ G, int ld,
                                           int row0, int k0, int tid,
                                           StageRegs& R) {
#pragma unroll
  for (int i = 0; i < 4; ++i) {
    const int lin = tid + (i << 8);
    const int r   = lin >> 3;                // 8 float4 per 32-wide row
    const int c   = (lin & 7) << 2;
    R.v[i] = *(const float4*)(G + (size_t)(row0 + r) * ld + k0 + c);
  }
}

__device__ __forceinline__ void stage_commit(const StageRegs& R, __bf16* S,
                                             int tid) {
#pragma unroll
  for (int i = 0; i < 4; ++i) {
    const int lin = tid + (i << 8);
    const int r   = lin >> 3;
    const int c   = (lin & 7) << 2;
    union { __bf16 h[4]; uint2 u2; } t;
    t.h[0] = (__bf16)R.v[i].x; t.h[1] = (__bf16)R.v[i].y;
    t.h[2] = (__bf16)R.v[i].z; t.h[3] = (__bf16)R.v[i].w;
    *(uint2*)(S + r * BK + c) = t.u2;
  }
}

// bf16 tile: copy global -> LDS directly with the CDNA5 async-to-LDS path
// (no VGPR round trip, tracked by ASYNCcnt). 2 x 16B per thread.
__device__ __forceinline__ void stage_bf16_async(const __bf16* __restrict__ G,
                                                 int ld, int row0, int k0,
                                                 __bf16* S, int tid) {
#pragma unroll
  for (int i = 0; i < 2; ++i) {
    const int lin = tid + (i << 8);
    const int r   = lin >> 2;                // 4 x 16B per 32-wide row
    const int c   = (lin & 3) << 3;
    const void* gp = (const void*)(G + (size_t)(row0 + r) * ld + k0 + c);
    // LDS aperture: low 32 bits of the generic address are the LDS byte addr.
    const uint32_t la = (uint32_t)(uintptr_t)(S + r * BK + c);
    asm volatile("global_load_async_to_lds_b128 %0, %1, off"
                 :: "v"(la), "v"(gp) : "memory");
  }
}

__device__ __forceinline__ void wait_async_lds() {
  asm volatile("s_wait_asynccnt 0x0" ::: "memory");
}

// ---------------------------------------------------------------------------
// Kernel 1/2: C[B,256](bf16) = X[B,K](f32) * W[256,K](f32)^T
// 128x128 block tile, 8 waves in 4(M)x2(N), each wave 32x64.
// Double-buffered LDS, one barrier per K-step.
__global__ __launch_bounds__(256)
void fastgrnn_rank_gemm(const float* __restrict__ X, const float* __restrict__ W,
                        __bf16* __restrict__ C, int K) {
  __shared__ __align__(16) __bf16 As[2][BM * BK];
  __shared__ __align__(16) __bf16 Bs[2][BN * BK];
  const int tid  = threadIdx.x;
  const int lane = tid & 31;
  const int wid  = tid >> 5;
  const int wm   = (wid & 3) * 32;
  const int wn   = (wid >> 2) * 64;
  const int rowB = blockIdx.y * BM;
  const int colB = blockIdx.x * BN;

  v8f zero = {};
  v8f acc[2][4];
#pragma unroll
  for (int i = 0; i < 2; ++i)
#pragma unroll
    for (int j = 0; j < 4; ++j) acc[i][j] = zero;

  StageRegs ra, rb;
  stage_load(X, K, rowB, 0, tid, ra);
  stage_load(W, K, colB, 0, tid, rb);
  stage_commit(ra, As[0], tid);
  stage_commit(rb, Bs[0], tid);

  const int nsteps = K / BK;
  for (int s = 0; s < nsteps; ++s) {
    __syncthreads();
    const int  cur  = s & 1;
    const bool more = (s + 1) < nsteps;
    if (more) {                               // issue next-tile global loads
      stage_load(X, K, rowB, (s + 1) * BK, tid, ra);
      stage_load(W, K, colB, (s + 1) * BK, tid, rb);
    }
    const v16bf a0 = frag_a(As[cur], wm, lane);
    const v16bf a1 = frag_a(As[cur], wm + 16, lane);
#pragma unroll
    for (int nt = 0; nt < 4; ++nt) {
      const v16bf b = frag_b(Bs[cur], wn + nt * 16, lane);
      acc[0][nt] = __builtin_amdgcn_wmma_f32_16x16x32_bf16(
          false, a0, false, b, (short)0, acc[0][nt], false, false);
      acc[1][nt] = __builtin_amdgcn_wmma_f32_16x16x32_bf16(
          false, a1, false, b, (short)0, acc[1][nt], false, false);
    }
    if (more) {                               // convert + park in other buffer
      stage_commit(ra, As[cur ^ 1], tid);
      stage_commit(rb, Bs[cur ^ 1], tid);
    }
  }

  // C/D layout: VGPR r -> M = r + 8*(lane/16), N = lane%16
  const int nloc = lane & 15;
  const int rg   = (lane >> 4) * 8;
#pragma unroll
  for (int mt = 0; mt < 2; ++mt)
#pragma unroll
    for (int nt = 0; nt < 4; ++nt) {
      const int col  = colB + wn + nt * 16 + nloc;
      const int row0 = rowB + wm + mt * 16 + rg;
#pragma unroll
      for (int r = 0; r < 8; ++r)
        C[(size_t)(row0 + r) * 256 + col] = (__bf16)acc[mt][nt][r];
    }
}

// ---------------------------------------------------------------------------
// Kernel 3: pre = C1*W2^T + C2*U2^T (flattened 16-step K pipeline over both
// rank-space operands into the same accumulators), then gate epilogue.
__global__ __launch_bounds__(256)
void fastgrnn_fused_gemm(const __bf16* __restrict__ C1, const __bf16* __restrict__ C2,
                         const float* __restrict__ W2, const float* __restrict__ U2,
                         const float* __restrict__ state,
                         const float* __restrict__ bias_g,
                         const float* __restrict__ bias_u,
                         const float* __restrict__ zeta,
                         const float* __restrict__ nu,
                         float* __restrict__ out) {
  __shared__ __align__(16) __bf16 As[2][BM * BK];
  __shared__ __align__(16) __bf16 Ws[2][BN * BK];
  const int tid  = threadIdx.x;
  const int lane = tid & 31;
  const int wid  = tid >> 5;
  const int wm   = (wid & 3) * 32;
  const int wn   = (wid >> 2) * 64;
  const int rowB = blockIdx.y * BM;
  const int colB = blockIdx.x * BN;

  const __bf16* Asrc[2] = {C1, C2};
  const float*  Wsrc[2] = {W2, U2};

  v8f zero = {};
  v8f acc[2][4];
#pragma unroll
  for (int i = 0; i < 2; ++i)
#pragma unroll
    for (int j = 0; j < 4; ++j) acc[i][j] = zero;

  StageRegs rw;
  stage_bf16_async(Asrc[0], 256, rowB, 0, As[0], tid);
  stage_load(Wsrc[0], 256, colB, 0, tid, rw);
  stage_commit(rw, Ws[0], tid);
  wait_async_lds();

  const int nsteps = 16;                      // 2 operands x (256/32)
  for (int s = 0; s < nsteps; ++s) {
    __syncthreads();
    const int  cur  = s & 1;
    const bool more = (s + 1) < nsteps;
    if (more) {
      const int t  = s + 1;
      const int p  = t >> 3;
      const int k0 = (t & 7) * BK;
      stage_bf16_async(Asrc[p], 256, rowB, k0, As[cur ^ 1], tid);
      stage_load(Wsrc[p], 256, colB, k0, tid, rw);
    }
    const v16bf a0 = frag_a(As[cur], wm, lane);
    const v16bf a1 = frag_a(As[cur], wm + 16, lane);
#pragma unroll
    for (int nt = 0; nt < 4; ++nt) {
      const v16bf b = frag_b(Ws[cur], wn + nt * 16, lane);
      acc[0][nt] = __builtin_amdgcn_wmma_f32_16x16x32_bf16(
          false, a0, false, b, (short)0, acc[0][nt], false, false);
      acc[1][nt] = __builtin_amdgcn_wmma_f32_16x16x32_bf16(
          false, a1, false, b, (short)0, acc[1][nt], false, false);
    }
    if (more) stage_commit(rw, Ws[cur ^ 1], tid);
    wait_async_lds();                         // async LDS writes done pre-barrier
  }

  const float sz = 1.f / (1.f + __expf(-zeta[0]));
  const float sn = 1.f / (1.f + __expf(-nu[0]));
  const int nloc = lane & 15;
  const int rg   = (lane >> 4) * 8;
#pragma unroll
  for (int mt = 0; mt < 2; ++mt)
#pragma unroll
    for (int nt = 0; nt < 4; ++nt) {
      const int col  = colB + wn + nt * 16 + nloc;
      const int row0 = rowB + wm + mt * 16 + rg;
      const float bg = bias_g[col];
      const float bu = bias_u[col];
#pragma unroll
      for (int r = 0; r < 8; ++r) {
        const float pre = acc[mt][nt][r];
        const float z   = 1.f / (1.f + __expf(-(pre + bg)));
        const float hc  = tanhf(pre + bu);
        const size_t idx = (size_t)(row0 + r) * 2048 + col;
        out[idx] = z * state[idx] + (sz * (1.f - z) + sn) * hc;
      }
    }
}

// ---------------------------------------------------------------------------
extern "C" void kernel_launch(void* const* d_in, const int* in_sizes, int n_in,
                              void* d_out, int out_size, void* d_ws, size_t ws_size,
                              hipStream_t stream) {
  (void)in_sizes; (void)n_in; (void)out_size; (void)ws_size;
  const float* x  = (const float*)d_in[0];   // [8192,1024]
  const float* h  = (const float*)d_in[1];   // [8192,2048]
  const float* W1 = (const float*)d_in[2];   // [256,1024]
  const float* W2 = (const float*)d_in[3];   // [2048,256]
  const float* U1 = (const float*)d_in[4];   // [256,2048]
  const float* U2 = (const float*)d_in[5];   // [2048,256]
  const float* bg = (const float*)d_in[6];   // [1,2048]
  const float* bu = (const float*)d_in[7];   // [1,2048]
  const float* zt = (const float*)d_in[8];   // [1,1]
  const float* nu = (const float*)d_in[9];   // [1,1]
  float* out = (float*)d_out;                // [8192,2048]

  __bf16* C1 = (__bf16*)d_ws;                   // 8192x256 bf16 (4 MB)
  __bf16* C2 = C1 + (size_t)8192 * 256;         // 8192x256 bf16 (4 MB)

  const dim3 blk(256);
  fastgrnn_rank_gemm<<<dim3(256 / BN, 8192 / BM), blk, 0, stream>>>(x, W1, C1, 1024);
  fastgrnn_rank_gemm<<<dim3(256 / BN, 8192 / BM), blk, 0, stream>>>(h, U1, C2, 2048);
  fastgrnn_fused_gemm<<<dim3(2048 / BN, 8192 / BM), blk, 0, stream>>>(
      C1, C2, W2, U2, h, bg, bu, zt, nu, out);
}
